// DynamicDictionaryLearning_27822798143569
// MI455X (gfx1250) — matrix operand order repaired and checked by reading, not compiled, not run on hardware
//
#include <hip/hip_runtime.h>
#include <hip/hip_bf16.h>

// ---------------------------------------------------------------------------
// DynamicDictionaryLearning on MI455X (gfx1250)
//   E = basic_queries(1024x2048) @ W_mlp(2048x16384) + b      [GEMM, bf16x3 WMMA]
//   dynamic[b,t,d] = sum_r qw[b,t,r] * E[t, r*2048+d]         [bandwidth kernel]
//   basic_expanded = broadcast(basic_queries)                  [fused into mix]
//
// Workspace layout (needs ~200 MiB):
//   [0,4MiB)        A_hi  bf16 fragments
//   [4,8MiB)        A_lo
//   [8,72MiB)       W_hi
//   [72,136MiB)     W_lo
//   [136,200MiB)    E     f32 (T x R*D) -- stays resident in 192MB L2
// ---------------------------------------------------------------------------

#define D_DIM 2048
#define T_DIM 1024
#define R_DIM 8
#define B_DIM 32
#define M_DIM T_DIM            // 1024
#define K_DIM D_DIM            // 2048
#define N_DIM (R_DIM * D_DIM)  // 16384
#define KT_N (K_DIM / 32)      // 64 k-tiles (K=32 per WMMA)
#define MT_N (M_DIM / 16)      // 64 m-tiles
#define NT_N (N_DIM / 16)      // 1024 n-tiles

typedef __attribute__((ext_vector_type(16))) __bf16 v16bf;
typedef __attribute__((ext_vector_type(8)))  float  v8f;
typedef __attribute__((ext_vector_type(4)))  float  v4f;

// ---------------------------------------------------------------------------
// Pack A (basic_queries) into per-lane WMMA A-fragment layout, split hi/lo bf16.
// ISA 16-bit A 16x32 layout: lane = (m%16) + 16*half,
//   k_local = 16*c + 8*half + 2*jj + b, element e = 2*(4c+jj) + b.
// ---------------------------------------------------------------------------
__global__ void __launch_bounds__(256) pack_a_kernel(
    const float* __restrict__ A, unsigned short* __restrict__ hi_raw,
    unsigned short* __restrict__ lo_raw) {
  __bf16* hi = (__bf16*)hi_raw;
  __bf16* lo = (__bf16*)lo_raw;
  int id = blockIdx.x * 256 + threadIdx.x;        // 0 .. M*K-1
  int m  = id >> 11;                              // / 2048
  int k  = id & (K_DIM - 1);
  float a  = A[id];
  __bf16 h = (__bf16)a;
  __bf16 l = (__bf16)(a - (float)h);
  int mt = m >> 4;
  int kt = k >> 5;
  int kl = k & 31;
  int c  = (kl >> 4) & 1;
  int hf = (kl >> 3) & 1;
  int jj = (kl >> 1) & 3;
  int e  = ((4 * c + jj) << 1) | (kl & 1);
  int lane = (m & 15) + (hf << 4);
  int dst  = (((mt * KT_N + kt) * 32 + lane) << 4) + e;
  hi[dst] = h;
  lo[dst] = l;
}

// ---------------------------------------------------------------------------
// Pack W_mlp into per-lane WMMA B-fragment layout, split hi/lo bf16.
// ISA 16-bit B 32x16 layout: lane = (n%16) + 16*((k_local>>3)&1), e = 8*(k_local>=16) + k_local%8.
// Fragment order [nt][kt] so the GEMM k-loop streams contiguously.
// ---------------------------------------------------------------------------
__global__ void __launch_bounds__(256) pack_w_kernel(
    const float* __restrict__ W, unsigned short* __restrict__ hi_raw,
    unsigned short* __restrict__ lo_raw) {
  __bf16* hi = (__bf16*)hi_raw;
  __bf16* lo = (__bf16*)lo_raw;
  int id = blockIdx.x * 256 + threadIdx.x;        // 0 .. K*N-1  (33.5M, fits int)
  int k  = id >> 14;                              // / 16384
  int n  = id & (N_DIM - 1);
  float a  = W[id];
  __bf16 h = (__bf16)a;
  __bf16 l = (__bf16)(a - (float)h);
  int nt = n >> 4;
  int kt = k >> 5;
  int kl = k & 31;
  int lane = (n & 15) + (((kl >> 3) & 1) << 4);
  int e    = ((kl >> 4) << 3) | (kl & 7);
  int dst  = (((nt * KT_N + kt) * 32 + lane) << 4) + e;
  hi[dst] = h;
  lo[dst] = l;
}

// ---------------------------------------------------------------------------
// GEMM: E = A @ W + bias via v_wmma_f32_16x16x32_bf16, bf16x3 split-precision.
// Block = 256 threads (8 waves): 4 m-waves x 2 n-waves, block tile 128x128.
// Each wave: 2 m-tiles x 4 n-tiles = 32x64 output, 64 accumulator VGPRs.
// ---------------------------------------------------------------------------
__global__ void __launch_bounds__(256, 1) gemm_bf16x3_kernel(
    const unsigned short* __restrict__ ahi_raw, const unsigned short* __restrict__ alo_raw,
    const unsigned short* __restrict__ whi_raw, const unsigned short* __restrict__ wlo_raw,
    const float* __restrict__ bias, float* __restrict__ E) {
  const v16bf* Ah = (const v16bf*)ahi_raw;
  const v16bf* Al = (const v16bf*)alo_raw;
  const v16bf* Wh = (const v16bf*)whi_raw;
  const v16bf* Wl = (const v16bf*)wlo_raw;

  const int lane = threadIdx.x & 31;
  const int w    = threadIdx.x >> 5;   // wave 0..7
  const int wm   = w & 3;              // 4 m-waves
  const int wn   = w >> 2;             // 2 n-waves
  const int mt0  = blockIdx.y * 8 + wm * 2;
  const int nt0  = blockIdx.x * 8 + wn * 4;

  v8f acc[2][4];
  const v8f vzero = {0.f, 0.f, 0.f, 0.f, 0.f, 0.f, 0.f, 0.f};
#pragma unroll
  for (int i = 0; i < 2; ++i)
#pragma unroll
    for (int n = 0; n < 4; ++n) acc[i][n] = vzero;

#pragma unroll 1
  for (int kt = 0; kt < KT_N; ++kt) {
    v16bf ah[2], al[2], wh[4], wl[4];
#pragma unroll
    for (int i = 0; i < 2; ++i) {
      int idx = ((mt0 + i) * KT_N + kt) * 32 + lane;
      ah[i] = Ah[idx];
      al[i] = Al[idx];
    }
#pragma unroll
    for (int n = 0; n < 4; ++n) {
      int idx = ((nt0 + n) * KT_N + kt) * 32 + lane;
      wh[n] = Wh[idx];
      wl[n] = Wl[idx];
      // stream-prefetch next k-tile of the big W operand (global_prefetch_b8)
      if (kt + 1 < KT_N) {
        __builtin_prefetch((const void*)(Wh + idx + 32), 0, 0);
        __builtin_prefetch((const void*)(Wl + idx + 32), 0, 0);
      }
    }
#pragma unroll
    for (int i = 0; i < 2; ++i)
#pragma unroll
      for (int n = 0; n < 4; ++n) {
        // E += Ahi*Whi + Ahi*Wlo + Alo*Whi  (f32 accumulate)
        acc[i][n] = __builtin_amdgcn_wmma_f32_16x16x32_bf16(
            false, ah[i], false, wh[n], (short)0, acc[i][n], false, false);
        acc[i][n] = __builtin_amdgcn_wmma_f32_16x16x32_bf16(
            false, ah[i], false, wl[n], (short)0, acc[i][n], false, false);
        acc[i][n] = __builtin_amdgcn_wmma_f32_16x16x32_bf16(
            false, al[i], false, wh[n], (short)0, acc[i][n], false, false);
      }
  }

  // Epilogue: C/D layout -> lane holds column n = lane%16; VGPR j holds row
  // m = j + 8*(lane>=16). Add bias per column, store f32 E row-major.
  const int col16 = lane & 15;
  const int rbase = (lane >> 4) * 8;
#pragma unroll
  for (int i = 0; i < 2; ++i) {
#pragma unroll
    for (int n = 0; n < 4; ++n) {
      int colg  = (nt0 + n) * 16 + col16;
      float bv  = bias[colg];
      int  row0 = (mt0 + i) * 16 + rbase;
#pragma unroll
      for (int j = 0; j < 8; ++j) {
        E[(size_t)(row0 + j) * N_DIM + colg] = acc[i][n][j] + bv;
      }
    }
  }
}

// ---------------------------------------------------------------------------
// Mix: dynamic[b,t,d] = sum_r qw[b,t,r]*E[t, r*D+d]; also writes basic_expanded.
// One thread per float4 of output; E is L2-resident (64 MiB < 192 MB L2),
// so HBM traffic is dominated by the mandatory 536 MB of stores.
// ---------------------------------------------------------------------------
__global__ void __launch_bounds__(256) mix_kernel(
    const float* __restrict__ qw, const float* __restrict__ bq,
    const float* __restrict__ E, float* __restrict__ out) {
  int id  = blockIdx.x * 256 + threadIdx.x;   // per float4, 16,777,216 total
  int g   = id << 2;                          // element offset in (B,T,D)
  int b   = g >> 21;                          // T*D = 2^21
  int rem = g & ((1 << 21) - 1);
  int t   = rem >> 11;                        // D = 2^11
  int d   = rem & (D_DIM - 1);

  const float* qp = qw + ((size_t)(b * T_DIM + t) << 3);  // 8 contiguous weights
  const v4f* Ev = (const v4f*)(E + (size_t)t * N_DIM + d);
  v4f a = {0.f, 0.f, 0.f, 0.f};
#pragma unroll
  for (int r = 0; r < R_DIM; ++r) {
    a += Ev[r * (D_DIM / 4)] * qp[r];
  }
  v4f* o = (v4f*)out;
  o[id] = a;                                             // dynamic_queries
  v4f bb = *(const v4f*)(bq + ((size_t)t << 11) + d);
  o[(size_t)(B_DIM * T_DIM * D_DIM / 4) + id] = bb;      // basic_expanded
}

// ---------------------------------------------------------------------------
extern "C" void kernel_launch(void* const* d_in, const int* in_sizes, int n_in,
                              void* d_out, int out_size, void* d_ws, size_t ws_size,
                              hipStream_t stream) {
  const float* qw = (const float*)d_in[0];   // (B,T,R)
  const float* bq = (const float*)d_in[1];   // (T,D)
  const float* Wm = (const float*)d_in[2];   // (D, R*D)
  const float* bm = (const float*)d_in[3];   // (R*D,)
  float* out = (float*)d_out;
  char* ws = (char*)d_ws;

  const size_t A_elems = (size_t)M_DIM * K_DIM;          // 2,097,152
  const size_t W_elems = (size_t)K_DIM * N_DIM;          // 33,554,432
  unsigned short* Ahi = (unsigned short*)ws;
  unsigned short* Alo = (unsigned short*)(ws + A_elems * 2);
  unsigned short* Whi = (unsigned short*)(ws + A_elems * 4);
  unsigned short* Wlo = (unsigned short*)(ws + A_elems * 4 + W_elems * 2);
  float*          E   = (float*)        (ws + A_elems * 4 + W_elems * 4);

  pack_a_kernel<<<(int)(A_elems / 256), 256, 0, stream>>>(bq, Ahi, Alo);
  pack_w_kernel<<<(int)(W_elems / 256), 256, 0, stream>>>(Wm, Whi, Wlo);

  dim3 ggrid(NT_N / 8, MT_N / 8, 1);   // (128, 8): 128-col x 128-row block tiles
  gemm_bf16x3_kernel<<<ggrid, 256, 0, stream>>>(Ahi, Alo, Whi, Wlo, bm, E);

  mix_kernel<<<(B_DIM * T_DIM * D_DIM / 4) / 256, 256, 0, stream>>>(qw, bq, E, out);
}
